// ExtendedMamba_32006096290220
// MI455X (gfx1250) — matrix operand order repaired and checked by reading, not compiled, not run on hardware
//
#include <hip/hip_runtime.h>
#include <math.h>

// ---------------------------------------------------------------------------
// VMamba / VSS forward for MI455X (gfx1250, wave32).
//   - GEMMs: v_wmma_f32_16x16x32_f16; operands pre-packed to f16 so each
//     fragment is global_load_b128; each wave computes a 16x32 strip (one A
//     fragment amortized over two WMMAs); epilogue = 2x b128 store per tile.
//   - Scan: sequential recurrence; coefficients staged into LDS in 48x64
//     tiles by the Tensor Data Mover (tensor_load_to_lds + s_wait_tensorcnt).
// ---------------------------------------------------------------------------

typedef __attribute__((ext_vector_type(16))) _Float16 v16h;
typedef __attribute__((ext_vector_type(8)))  float    v8f;
typedef __attribute__((ext_vector_type(4)))  float    v4f;
typedef __attribute__((ext_vector_type(4)))  int      i32x4;
typedef __attribute__((ext_vector_type(8)))  int      i32x8;
typedef __attribute__((ext_vector_type(4)))  unsigned u32x4;

constexpr int Bc  = 2;
constexpr int Lc  = 4096;
constexpr int DMc = 128;
constexpr int DIc = 256;             // 2 * DM
constexpr int Kc  = 4;               // scan directions
constexpr int Nc  = 16;              // d_state
constexpr int DRc = 8;               // dt_rank
constexpr int DCc = 4;               // d_conv
constexpr int HWc = 64;              // H == W
constexpr int CRAW  = DRc + 2 * Nc;  // 40 live rows of x_dbl
constexpr int CROWS = 48;            // padded to 3x16 for WMMA tiling
constexpr int CHUNK = 64;            // scan steps staged per TDM transfer

#if defined(__has_builtin)
#if __has_builtin(__builtin_amdgcn_tensor_load_to_lds)
#define HAVE_TDM 1
#endif
#endif

// cross-scan index map: direction k, scan-time index l -> original l
__device__ __forceinline__ int dirmap(int k, int l) {
  int ll = (k & 2) ? (Lc - 1 - l) : l;                 // reversed directions
  if (k & 1) ll = ((ll & (HWc - 1)) << 6) | (ll >> 6); // 64x64 transpose
  return ll;
}

// ---------------------------------------------------------------------------
// One wave computes a 16(M) x 32(N) strip of A(16xKD) * B(KDx32).
// A: f16 row-major (K contiguous). B: f16 with K contiguous per column
// (activations stored (l, d)). Per CDNA5 7.12.2 each lane fragment is two
// contiguous 16B runs -> all operand traffic is global_load_b128.
// ---------------------------------------------------------------------------
struct Acc2 { v8f a0, a1; };

template <int KD>
__device__ __forceinline__ Acc2 wmma_strip(const _Float16* __restrict__ A, int lda,
                                           const _Float16* __restrict__ B, int ldb) {
  const int lane = threadIdx.x & 31;
  const int hi   = lane >> 4;
  const int r    = lane & 15;
  const _Float16* ar  = A + (size_t)r * lda + 8 * hi;   // A: M=r
  const _Float16* br0 = B + (size_t)r * ldb + 8 * hi;   // B cols n0..n0+15
  const _Float16* br1 = br0 + (size_t)16 * ldb;         // B cols n0+16..+31
  v8f c0 = {}, c1 = {};
#pragma unroll
  for (int k0 = 0; k0 < KD; k0 += 32) {
    v16h a, b0, b1;
    i32x4* ap  = reinterpret_cast<i32x4*>(&a);
    i32x4* bp0 = reinterpret_cast<i32x4*>(&b0);
    i32x4* bp1 = reinterpret_cast<i32x4*>(&b1);
    ap[0]  = *(const i32x4*)(ar + k0);        // A elems 0-7 : K=k0+8hi+0..7
    ap[1]  = *(const i32x4*)(ar + k0 + 16);   // A elems 8-15: K=k0+16+8hi+..
    bp0[0] = *(const i32x4*)(br0 + k0);       // B elems 0-7 : K=k0+8hi+0..7
    bp0[1] = *(const i32x4*)(br0 + k0 + 16);  // B elems 8-15: K=k0+16+8hi+..
    bp1[0] = *(const i32x4*)(br1 + k0);
    bp1[1] = *(const i32x4*)(br1 + k0 + 16);
    c0 = __builtin_amdgcn_wmma_f32_16x16x32_f16(false, a, false, b0, (short)0,
                                                c0, false, false);
    c1 = __builtin_amdgcn_wmma_f32_16x16x32_f16(false, a, false, b1, (short)0,
                                                c1, false, false);
  }
  return {c0, c1};
}

// Store one 16x16 f32 tile: this lane's 8 values are contiguous -> 2x b128.
__device__ __forceinline__ void store_tile(float* rowptr, const v8f& acc) {
  const int lane = threadIdx.x & 31, hi = lane >> 4;
  float* p = rowptr + 8 * hi;
  *(v4f*)(p)     = __builtin_shufflevector(acc, acc, 0, 1, 2, 3);
  *(v4f*)(p + 4) = __builtin_shufflevector(acc, acc, 4, 5, 6, 7);
}

// ------------------------- pack kernels ------------------------------------
__global__ void k_f32_to_f16(const float* __restrict__ s,
                             _Float16* __restrict__ d, int n) {
  const int i = blockIdx.x * blockDim.x + threadIdx.x;
  if (i < n) d[i] = (_Float16)s[i];
}

// x_proj_w (K,40,DI) -> padded f16 (K,48,DI)
__global__ void k_pack_xw(const float* __restrict__ xw,
                          _Float16* __restrict__ w16x) {
  const int i = blockIdx.x * blockDim.x + threadIdx.x;
  if (i >= Kc * CROWS * DIc) return;
  const int d = i % DIc, c = (i / DIc) % CROWS, k = i / (DIc * CROWS);
  w16x[i] = (c < CRAW) ? (_Float16)xw[((size_t)k * CRAW + c) * DIc + d]
                       : (_Float16)0.f;
}

// xs16[b,k,l,d] = f16(xconv[b, dirmap(k,l), d])  (both sides d-coalesced)
__global__ void k_pack_xs(const float* __restrict__ xc32,
                          _Float16* __restrict__ xs16) {
  const int i = blockIdx.x * blockDim.x + threadIdx.x;
  if (i >= Bc * Kc * Lc * DIc) return;
  const int d = i % DIc;
  const int l = (i / DIc) % Lc;
  const int k = (i / (DIc * Lc)) % Kc;
  const int b = i / (DIc * Lc * Kc);
  xs16[i] = (_Float16)xc32[((size_t)b * Lc + dirmap(k, l)) * DIc + d];
}

// ------------------------- 1) in_proj GEMM ---------------------------------
// xz[b,e,l] = sum_d hs[b,l,d] * W[e,d]; x plane and z plane in (b,l,ch).
__global__ void k_inproj(const _Float16* __restrict__ hs16,
                         const _Float16* __restrict__ w16,
                         float* __restrict__ xraw, float* __restrict__ zbuf) {
  const int TM = (2 * DIc) / 16, TN = Lc / 32;
  const int tile = blockIdx.x * (blockDim.x >> 5) + (threadIdx.x >> 5);
  if (tile >= Bc * TM * TN) return;
  const int b  = tile / (TM * TN);
  const int rm = tile % (TM * TN);
  const int e0 = (rm / TN) * 16, l0 = (rm % TN) * 32;

  Acc2 acc = wmma_strip<DMc>(w16 + (size_t)e0 * DMc, DMc,
                             hs16 + ((size_t)b * Lc + l0) * DMc, DMc);

  const int r = threadIdx.x & 15;
  // tile-uniform x/z split (e0 multiple of 16, DIc multiple of 16)
  float* plane = (e0 < DIc) ? xraw : zbuf;
  const int eb = (e0 < DIc) ? e0 : e0 - DIc;
  store_tile(plane + ((size_t)b * Lc + l0 + r) * DIc + eb, acc.a0);
  store_tile(plane + ((size_t)b * Lc + l0 + 16 + r) * DIc + eb, acc.a1);
}

// ------------------------- 2) depthwise conv + SiLU ------------------------
__global__ void k_conv_silu(const float* __restrict__ xraw,
                            const float* __restrict__ cw,
                            const float* __restrict__ cb,
                            float* __restrict__ xc32) {
  const int i = blockIdx.x * blockDim.x + threadIdx.x;
  if (i >= Bc * Lc * DIc) return;
  const int d = i % DIc;
  const int l = (i / DIc) % Lc;
  const int b = i / (DIc * Lc);
  float acc = cb[d];
#pragma unroll
  for (int j = 0; j < DCc; ++j) {
    const int ll = l - (DCc - 1) + j;
    if (ll >= 0) acc += cw[d * DCc + j] * xraw[((size_t)b * Lc + ll) * DIc + d];
  }
  xc32[i] = acc / (1.f + __expf(-acc));   // SiLU
}

// ------------------------- 3) x_proj GEMM (4 directions) -------------------
// xdbl[b,k,l,c] = sum_d xs[b,k,l,d] * x_proj_w[k,c,d]
__global__ void k_xproj(const _Float16* __restrict__ xs16,
                        const _Float16* __restrict__ w16x,
                        float* __restrict__ xdbl) {
  const int TM = CROWS / 16, TN = Lc / 32;
  const int tile = blockIdx.x * (blockDim.x >> 5) + (threadIdx.x >> 5);
  if (tile >= Bc * Kc * TM * TN) return;
  int t = tile;
  const int b = t / (Kc * TM * TN); t %= Kc * TM * TN;
  const int k = t / (TM * TN);      t %= TM * TN;
  const int c0 = (t / TN) * 16, l0 = (t % TN) * 32;

  Acc2 acc = wmma_strip<DIc>(w16x + ((size_t)k * CROWS + c0) * DIc, DIc,
                             xs16 + (((size_t)b * Kc + k) * Lc + l0) * DIc, DIc);

  const int r = threadIdx.x & 15;
  float* base = xdbl + ((size_t)b * Kc + k) * Lc * CROWS + c0;
  store_tile(base + (size_t)(l0 + r) * CROWS, acc.a0);
  store_tile(base + (size_t)(l0 + 16 + r) * CROWS, acc.a1);
}

// ------------------------- 4) selective scan -------------------------------
#ifdef HAVE_TDM
// Issue a 2D TDM transfer: 48x64 f32 tile (one CHUNK of coefficient columns)
// from xdbl (contiguous) into LDS. Executed by wave 0 only.
__device__ __forceinline__ void tdm_load_cols(const float* gptr, unsigned lds_off) {
  const unsigned long long ga = (unsigned long long)(uintptr_t)gptr;
  u32x4 g0;
  g0[0] = 1u;                                   // count=1 (valid descriptor)
  g0[1] = lds_off;                              // lds_addr (bytes)
  g0[2] = (unsigned)ga;                         // global_addr[31:0]
  g0[3] = (unsigned)(ga >> 32) | (2u << 30);    // global_addr[56:32] | type=2
  i32x8 g1;
  g1[0] = (int)(2u << 16);                      // data_size = 4 bytes
  g1[1] = (int)((unsigned)CROWS << 16);         // tensor_dim0 = 48
  g1[2] = (int)((unsigned)CHUNK << 16);         // tensor_dim1 = 64
  g1[3] = (int)((unsigned)CROWS << 16);         // tile_dim0 = 48
  g1[4] = CHUNK;                                // tile_dim1 = 64 (tile_dim2=0)
  g1[5] = CROWS;                                // tensor_dim0_stride = 48
  g1[6] = 0;
  g1[7] = 0;
  i32x4 gz4 = {0, 0, 0, 0};
#if __clang_major__ >= 23
  i32x8 gz8 = {0, 0, 0, 0, 0, 0, 0, 0};
  __builtin_amdgcn_tensor_load_to_lds(g0, g1, gz4, gz4, gz8, 0);
#else
  __builtin_amdgcn_tensor_load_to_lds(g0, g1, gz4, gz4, 0);
#endif
}
#endif

// One block per (b,k); thread = channel d; 16-state recurrence in registers.
// Fuses dt-rank projection + softplus; adds D*x; stores y as (B,K,L,DI).
__global__ void __launch_bounds__(DIc) k_scan(
    const float* __restrict__ xc32, const float* __restrict__ xdbl,
    const float* __restrict__ dtw, const float* __restrict__ dtb,
    const float* __restrict__ alogs, const float* __restrict__ ds,
    float* __restrict__ ydir) {
  const int b = blockIdx.x / Kc, k = blockIdx.x % Kc, d = threadIdx.x;
  const int kd = k * DIc + d;

  float wdt[DRc];
#pragma unroll
  for (int r = 0; r < DRc; ++r) wdt[r] = dtw[(size_t)kd * DRc + r];
  const float bias = dtb[kd];
  const float Dv   = ds[kd];
  float An[Nc], h[Nc];
#pragma unroll
  for (int n = 0; n < Nc; ++n) {
    An[n] = -__expf(alogs[(size_t)kd * Nc + n]);  // A = -exp(A_log)
    h[n] = 0.f;
  }

  __shared__ float sm[CHUNK * CROWS];             // 12 KB coefficient stage
  const float* colbase = xdbl + ((size_t)b * Kc + k) * Lc * CROWS;
  float*       yout    = ydir + ((size_t)b * Kc + k) * (size_t)Lc * DIc;
  const float* xcb     = xc32 + (size_t)b * Lc * DIc;

  for (int c0 = 0; c0 < Lc; c0 += CHUNK) {
#ifdef HAVE_TDM
    if (threadIdx.x < 32) {                       // wave 0 drives the TDM
      tdm_load_cols(colbase + (size_t)c0 * CROWS,
                    (unsigned)(uintptr_t)(void*)sm);
      __builtin_amdgcn_s_wait_tensorcnt(0);
    }
#else
    for (int j = threadIdx.x; j < CHUNK * CROWS; j += DIc)
      sm[j] = colbase[(size_t)c0 * CROWS + j];
#endif
    __syncthreads();

    for (int i = 0; i < CHUNK; ++i) {
      const int l = c0 + i;
      const float* row = sm + i * CROWS;          // broadcast reads
      float draw = bias;
#pragma unroll
      for (int r = 0; r < DRc; ++r) draw += row[r] * wdt[r];
      const float delta = (draw > 20.f) ? draw : __logf(1.f + __expf(draw));

      const float x  = xcb[(size_t)dirmap(k, l) * DIc + d];
      const float dx = delta * x;
      float y = 0.f;
#pragma unroll
      for (int n = 0; n < Nc; ++n) {
        h[n] = __expf(delta * An[n]) * h[n] + dx * row[DRc + n];
        y += h[n] * row[DRc + Nc + n];
      }
      y += Dv * x;
      yout[(size_t)l * DIc + d] = y;

      if (l + 1 < Lc)                             // global_prefetch_b8
        __builtin_prefetch(&xcb[(size_t)dirmap(k, l + 1) * DIc + d], 0, 0);
    }
    __syncthreads();
  }
}

// ------------------------- 5) merge + LayerNorm + gate ---------------------
__global__ void __launch_bounds__(DIc) k_merge(
    const float* __restrict__ ydir, const float* __restrict__ zbuf,
    const float* __restrict__ lng, const float* __restrict__ lnb,
    _Float16* __restrict__ yf16) {
  const int b = blockIdx.x / Lc, l = blockIdx.x % Lc, d = threadIdx.x;
  const int lt = ((l & 63) << 6) | (l >> 6);       // transpose partner index
  const size_t base = (size_t)b * Kc * Lc * DIc;

  float y = ydir[base + ((size_t)0 * Lc + l) * DIc + d]
          + ydir[base + ((size_t)1 * Lc + lt) * DIc + d]
          + ydir[base + ((size_t)2 * Lc + (Lc - 1 - l)) * DIc + d]
          + ydir[base + ((size_t)3 * Lc + (Lc - 1 - lt)) * DIc + d];

  __shared__ float red[DIc];
  red[d] = y; __syncthreads();
  for (int s = DIc / 2; s > 0; s >>= 1) {
    if (d < s) red[d] += red[d + s];
    __syncthreads();
  }
  const float mu = red[0] * (1.f / DIc);
  __syncthreads();
  const float yc = y - mu;
  red[d] = yc * yc; __syncthreads();
  for (int s = DIc / 2; s > 0; s >>= 1) {
    if (d < s) red[d] += red[d + s];
    __syncthreads();
  }
  const float var = red[0] * (1.f / DIc);

  const float yn = yc * rsqrtf(var + 1e-5f) * lng[d] + lnb[d];
  const float zv = zbuf[((size_t)b * Lc + l) * DIc + d];
  yf16[((size_t)b * Lc + l) * DIc + d] =
      (_Float16)(yn * (zv / (1.f + __expf(-zv))));
}

// ------------------------- 6) out_proj GEMM --------------------------------
__global__ void k_outproj(const _Float16* __restrict__ yf16,
                          const _Float16* __restrict__ w16o,
                          float* __restrict__ out) {
  const int TM = DMc / 16, TN = Lc / 32;
  const int tile = blockIdx.x * (blockDim.x >> 5) + (threadIdx.x >> 5);
  if (tile >= Bc * TM * TN) return;
  int t = tile;
  const int b = t / (TM * TN); t %= TM * TN;
  const int m0 = (t / TN) * 16, l0 = (t % TN) * 32;

  Acc2 acc = wmma_strip<DIc>(w16o + (size_t)m0 * DIc, DIc,
                             yf16 + ((size_t)b * Lc + l0) * DIc, DIc);

  const int r = threadIdx.x & 15;
  float* base = out + (size_t)b * Lc * DMc + m0;
  store_tile(base + (size_t)(l0 + r) * DMc, acc.a0);
  store_tile(base + (size_t)(l0 + 16 + r) * DMc, acc.a1);
}

// ---------------------------------------------------------------------------
extern "C" void kernel_launch(void* const* d_in, const int* in_sizes, int n_in,
                              void* d_out, int out_size, void* d_ws,
                              size_t ws_size, hipStream_t stream) {
  const float* hs  = (const float*)d_in[0];   // (B,L,DM)
  const float* inw = (const float*)d_in[1];   // (2DI,DM)
  const float* cw  = (const float*)d_in[2];   // (DI,1,DC)
  const float* cb  = (const float*)d_in[3];   // (DI)
  const float* xw  = (const float*)d_in[4];   // (K,40,DI)
  const float* dtw = (const float*)d_in[5];   // (K,DI,DR)
  const float* dtb = (const float*)d_in[6];   // (K*DI)
  const float* al  = (const float*)d_in[7];   // (K*DI,N)
  const float* ds  = (const float*)d_in[8];   // (K*DI)
  const float* lng = (const float*)d_in[9];   // (DI)
  const float* lnb = (const float*)d_in[10];  // (DI)
  const float* ow  = (const float*)d_in[11];  // (DM,DI)
  float* out = (float*)d_out;

  float* ws = (float*)d_ws;
  const size_t S1 = (size_t)Bc * Lc * DIc;          // activation plane (f32)
  const size_t S2 = (size_t)Bc * Kc * Lc * CROWS;   // xdbl (f32)
  const size_t S3 = (size_t)Bc * Kc * Lc * DIc;     // ydir (f32)
  float* xraw = ws;
  float* zbuf = ws + S1;
  float* xc32 = ws + 2 * S1;
  float* xdbl = ws + 3 * S1;
  float* ydir = ws + 3 * S1 + S2;
  float* fp   = ws + 3 * S1 + S2 + S3;              // f16 region (as floats)
  _Float16* hs16  = (_Float16*)fp;                          // B*L*DM
  _Float16* xs16  = (_Float16*)(fp + S1 / 4);               // B*K*L*DI
  _Float16* yf16  = (_Float16*)(fp + S1 / 4 + S3 / 2);      // B*L*DI
  _Float16* w16in = (_Float16*)(fp + S1 / 4 + S3 / 2 + S1 / 2);
  _Float16* w16x  = w16in + 2 * DIc * DMc;                  // K*48*DI
  _Float16* w16o  = w16x + Kc * CROWS * DIc;                // DM*DI

  // pack weights + input to f16
  { const int n = Bc * Lc * DMc;    k_f32_to_f16<<<(n + 255) / 256, 256, 0, stream>>>(hs, hs16, n); }
  { const int n = 2 * DIc * DMc;    k_f32_to_f16<<<(n + 255) / 256, 256, 0, stream>>>(inw, w16in, n); }
  { const int n = DMc * DIc;        k_f32_to_f16<<<(n + 255) / 256, 256, 0, stream>>>(ow, w16o, n); }
  { const int n = Kc * CROWS * DIc; k_pack_xw<<<(n + 255) / 256, 256, 0, stream>>>(xw, w16x); }

  { const int tiles = Bc * ((2 * DIc) / 16) * (Lc / 32);
    k_inproj<<<tiles / 8, 256, 0, stream>>>(hs16, w16in, xraw, zbuf); }
  { const int n = Bc * Lc * DIc;
    k_conv_silu<<<n / 256, 256, 0, stream>>>(xraw, cw, cb, xc32); }
  { const int n = Bc * Kc * Lc * DIc;
    k_pack_xs<<<n / 256, 256, 0, stream>>>(xc32, xs16); }
  { const int tiles = Bc * Kc * (CROWS / 16) * (Lc / 32);
    k_xproj<<<tiles / 8, 256, 0, stream>>>(xs16, w16x, xdbl); }
  k_scan<<<Bc * Kc, DIc, 0, stream>>>(xc32, xdbl, dtw, dtb, al, ds, ydir);
  k_merge<<<Bc * Lc, DIc, 0, stream>>>(ydir, zbuf, lng, lnb, yf16);
  { const int tiles = Bc * (DMc / 16) * (Lc / 32);
    k_outproj<<<tiles / 8, 256, 0, stream>>>(yf16, w16o, out); }

  (void)in_sizes; (void)n_in; (void)out_size; (void)ws_size;
}
